// GRUTrajectoryHead_57037165690906
// MI455X (gfx1250) — matrix operand order repaired and checked by reading, not compiled, not run on hardware
//
#include <hip/hip_runtime.h>
#include <math.h>

// Problem constants (match reference)
#define B_SZ 4096
#define D_SZ 1024
#define H_SZ 1024
#define T_SZ 50

typedef unsigned int u32;
typedef unsigned short u16;
typedef __attribute__((ext_vector_type(4)))  u32    u32x4;
typedef __attribute__((ext_vector_type(4)))  int    i32x4;
typedef __attribute__((ext_vector_type(8)))  float  v8f;
typedef __attribute__((ext_vector_type(16))) __bf16 v16bf;

struct Frag32B { u32x4 lo, hi; };

static __device__ __forceinline__ v16bf make_frag(u32x4 lo, u32x4 hi) {
    Frag32B f; f.lo = lo; f.hi = hi;
    return __builtin_bit_cast(v16bf, f);
}

// round-to-nearest-even fp32 -> bf16 (as raw u16)
static __device__ __forceinline__ u16 f2bf(float x) {
    u32 u = __builtin_bit_cast(u32, x);
    u += 0x7FFFu + ((u >> 16) & 1u);
    return (u16)(u >> 16);
}

static __device__ __forceinline__ float fast_sigmoid(float x) {
    x = fminf(15.f, fmaxf(-15.f, x));
    return 1.f / (1.f + __expf(-x));
}
static __device__ __forceinline__ float fast_tanh(float x) {
    x = fminf(15.f, fmaxf(-15.f, x));
    float e = __expf(2.f * x);
    return (e - 1.f) / (e + 1.f);
}

// ---- async global->LDS copy (GLOBAL_LOAD_ASYNC_TO_LDS_B128), guarded ----
#if __has_builtin(__builtin_amdgcn_global_load_async_to_lds_b128)
#define HAS_ASYNC_LDS 1
#else
#define HAS_ASYNC_LDS 0
#endif

typedef __attribute__((address_space(1))) i32x4 as1_i32x4;
typedef __attribute__((address_space(3))) i32x4 as3_i32x4;

static __device__ __forceinline__ void g2l_16B(const u16* g, u16* l) {
#if HAS_ASYNC_LDS
    __builtin_amdgcn_global_load_async_to_lds_b128(
        (as1_i32x4*)g, (as3_i32x4*)l, 0, 0);
#else
    *(u32x4*)l = *(const u32x4*)g;   // reg bounce; __syncthreads covers it
#endif
}

static __device__ __forceinline__ void wait_async0() {
#if __has_builtin(__builtin_amdgcn_s_wait_asynccnt)
    __builtin_amdgcn_s_wait_asynccnt(0);
#elif HAS_ASYNC_LDS
    asm volatile("s_wait_asynccnt 0x0" ::: "memory");
#endif
}

// ---------------------------------------------------------------- convert
__global__ void cvt_f32_bf16_kernel(const float* __restrict__ in,
                                    u16* __restrict__ out, int n) {
    int i = blockIdx.x * blockDim.x + threadIdx.x;
    if (i < n) out[i] = f2bf(in[i]);
}

// ---------------------------------------------------------------- h0 GEMM
// h0 = tanh(ev @ w_init^T + b_init)   (4096x1024 @ 1024x1024)
// block: 256 threads = 8 waves, 4(M) x 2(N) -> 64x32 block tile
__global__ __launch_bounds__(256) void h0_kernel(
    const u16*  __restrict__ evb,     // B x D bf16
    const u16*  __restrict__ wib,     // H x D bf16 (row-major, K contiguous)
    const float* __restrict__ b_init, // H
    float* __restrict__ hf, u16* __restrict__ hb) {
    const int lane = threadIdx.x & 31;
    const int wid  = threadIdx.x >> 5;
    const int m0 = blockIdx.x * 64 + (wid >> 1) * 16;
    const int n0 = blockIdx.y * 32 + (wid & 1) * 16;
    const int l16 = lane & 15, lh = lane >> 4;

    const u16* ap = evb + (size_t)(m0 + l16) * D_SZ + (lh << 3);
    const int  col = n0 + l16;
    const u16* bp = wib + (size_t)col * D_SZ + (lh << 3);

    v8f acc = {};
    for (int k0 = 0; k0 < D_SZ; k0 += 32) {
        v16bf a = make_frag(*(const u32x4*)(ap + k0), *(const u32x4*)(ap + k0 + 16));
        v16bf b = make_frag(*(const u32x4*)(bp + k0), *(const u32x4*)(bp + k0 + 16));
        acc = __builtin_amdgcn_wmma_f32_16x16x32_bf16(false, a, false, b,
                                                      (short)0, acc, false, false);
    }
    const float bi = b_init[col];
#pragma unroll
    for (int i = 0; i < 8; ++i) {
        int m = m0 + i + (lh << 3);
        float h = fast_tanh(acc[i] + bi);
        hf[(size_t)m * H_SZ + col] = h;
        hb[(size_t)m * H_SZ + col] = f2bf(h);
    }
}

// ---------------------------------------------------------------- GRU step
// Fused gh = h @ w_hh^T with gate epilogue.
// Block tile: 128(M) x 32(N); 8 waves as 4(M) x 2(N); wave tile 32x16 with
// 6 accumulators (r/z/n x 2 M-subtiles). The 96 w_hh rows (3 gates x 32
// cols) of each 64-wide K chunk are staged in LDS via async copies
// (double-buffered) and shared by all waves; A fragments come straight
// from global (L2-resident h state).
#define KC 64
#define ROWSTRIDE 72   // 64 u16 + 8 pad (144 B, 16B-aligned rows)
__global__ __launch_bounds__(256) void gru_step_kernel(
    const u16*   __restrict__ hb_prev,  // B x H bf16
    const float* __restrict__ hf_prev,  // B x H f32
    const u16*   __restrict__ whh,      // 3H x H bf16
    const float* __restrict__ b_hh,     // 3H
    const float* __restrict__ w_ih,     // 3H x 2
    const float* __restrict__ b_ih,     // 3H
    const float* __restrict__ teacher,  // B x T x 2
    int t,
    float* __restrict__ hf_new, u16* __restrict__ hb_new) {
    __shared__ __align__(16) u16 ldsB[2][96 * ROWSTRIDE];   // 2 x 13.5 KB

    const int tid  = threadIdx.x;
    const int lane = tid & 31;
    const int wid  = tid >> 5;
    const int wm   = wid >> 1;          // 0..3
    const int wn   = wid & 1;           // 0..1
    const int l16  = lane & 15, lh = lane >> 4;

    const int nblk = blockIdx.y * 32;          // first column of block
    const int m0w  = blockIdx.x * 128 + wm * 32;
    const int cin  = wn * 16 + l16;            // column within block (0..31)
    const int col  = nblk + cin;               // global column

    // A fragment base pointers (two 16-row M subtiles)
    const u16* ap0 = hb_prev + (size_t)(m0w + l16) * H_SZ + (lh << 3);
    const u16* ap1 = ap0 + (size_t)16 * H_SZ;

    // stage one 64-wide K chunk of all 96 weight rows into LDS buffer `buf`
    auto stage = [&](int buf, int k0) {
#pragma unroll
        for (int j = 0; j < 3; ++j) {
            int idx = tid + j * 256;        // 0..767
            int row = idx >> 3;             // 0..95  (= gate*32 + col-in-blk)
            int seg = idx & 7;              // 8 x 16B segments per row
            int g = row >> 5, c = row & 31;
            const u16* gp = whh + (size_t)(g * H_SZ + nblk + c) * H_SZ
                                + k0 + seg * 8;
            g2l_16B(gp, &ldsB[buf][row * ROWSTRIDE + seg * 8]);
        }
    };

    v8f accR[2] = {v8f{}, v8f{}};
    v8f accZ[2] = {v8f{}, v8f{}};
    v8f accN[2] = {v8f{}, v8f{}};

    stage(0, 0);
    int buf = 0;
    for (int kc = 0; kc < H_SZ / KC; ++kc) {
        wait_async0();        // my copies into `buf` are complete
        __syncthreads();      // everyone's copies done; prior reads retired
        if (kc + 1 < H_SZ / KC) stage(buf ^ 1, (kc + 1) * KC);

        const int k0 = kc * KC;
#pragma unroll
        for (int s = 0; s < 2; ++s) {
            const int k = k0 + s * 32;
            v16bf a0 = make_frag(*(const u32x4*)(ap0 + k),
                                 *(const u32x4*)(ap0 + k + 16));
            v16bf a1 = make_frag(*(const u32x4*)(ap1 + k),
                                 *(const u32x4*)(ap1 + k + 16));
#pragma unroll
            for (int g = 0; g < 3; ++g) {
                const u16* lb = &ldsB[buf][(g * 32 + cin) * ROWSTRIDE
                                           + s * 32 + (lh << 3)];
                v16bf b = make_frag(*(const u32x4*)lb, *(const u32x4*)(lb + 16));
                v8f* acc = (g == 0) ? accR : (g == 1) ? accZ : accN;
                acc[0] = __builtin_amdgcn_wmma_f32_16x16x32_bf16(
                    false, a0, false, b, (short)0, acc[0], false, false);
                acc[1] = __builtin_amdgcn_wmma_f32_16x16x32_bf16(
                    false, a1, false, b, (short)0, acc[1], false, false);
            }
        }
        buf ^= 1;
    }

    // per-column constants (one column per lane in the 16x16 D layout)
    const float bhr = b_hh[col];
    const float bhz = b_hh[H_SZ + col];
    const float bhn = b_hh[2 * H_SZ + col];
    const float wir0 = w_ih[(size_t)col * 2],              wir1 = w_ih[(size_t)col * 2 + 1];
    const float wiz0 = w_ih[(size_t)(H_SZ + col) * 2],     wiz1 = w_ih[(size_t)(H_SZ + col) * 2 + 1];
    const float win0 = w_ih[(size_t)(2 * H_SZ + col) * 2], win1 = w_ih[(size_t)(2 * H_SZ + col) * 2 + 1];
    const float bir = b_ih[col];
    const float biz = b_ih[H_SZ + col];
    const float bin = b_ih[2 * H_SZ + col];

#pragma unroll
    for (int mt = 0; mt < 2; ++mt) {
#pragma unroll
        for (int i = 0; i < 8; ++i) {
            int m = m0w + mt * 16 + i + (lh << 3);
            float p0 = 0.f, p1 = 0.f;
            if (t > 0) {  // uniform branch
                size_t to = ((size_t)m * T_SZ + (t - 1)) * 2;
                p0 = teacher[to];
                p1 = teacher[to + 1];
            }
            float ghr = accR[mt][i] + bhr;
            float ghz = accZ[mt][i] + bhz;
            float ghn = accN[mt][i] + bhn;
            float r  = fast_sigmoid(p0 * wir0 + p1 * wir1 + bir + ghr);
            float z  = fast_sigmoid(p0 * wiz0 + p1 * wiz1 + biz + ghz);
            float ng = fast_tanh   (p0 * win0 + p1 * win1 + bin + r * ghn);
            float hp = hf_prev[(size_t)m * H_SZ + col];
            float hv = (1.f - z) * ng + z * hp;
            hf_new[(size_t)m * H_SZ + col] = hv;
            hb_new[(size_t)m * H_SZ + col] = f2bf(hv);
        }
    }
}

// ---------------------------------------------------------------- delta
// delta[b] = h_new[b] @ w_out^T + b_out  (one wave per row, shuffle reduce)
__global__ __launch_bounds__(256) void delta_kernel(
    const float* __restrict__ hf, const float* __restrict__ w_out,
    const float* __restrict__ b_out, float* __restrict__ pd, int t) {
    const int lane = threadIdx.x & 31;
    const int b = blockIdx.x * 8 + (threadIdx.x >> 5);
    const float* hr = hf + (size_t)b * H_SZ;
    float s0 = 0.f, s1 = 0.f;
    for (int k = lane; k < H_SZ; k += 32) {
        float h = hr[k];
        s0 += h * w_out[k];
        s1 += h * w_out[H_SZ + k];
    }
#pragma unroll
    for (int off = 16; off; off >>= 1) {
        s0 += __shfl_down(s0, off, 32);
        s1 += __shfl_down(s1, off, 32);
    }
    if (lane == 0) {
        size_t o = ((size_t)b * T_SZ + t) * 2;
        pd[o]     = s0 + b_out[0];
        pd[o + 1] = s1 + b_out[1];
    }
}

// ---------------------------------------------------------------- cumsum
__global__ void cumsum_kernel(const float* __restrict__ pd, float* __restrict__ pp) {
    int idx = blockIdx.x * blockDim.x + threadIdx.x;  // b*2 + j
    if (idx >= B_SZ * 2) return;
    int b = idx >> 1, j = idx & 1;
    float acc = 0.f;
    for (int t = 0; t < T_SZ; ++t) {
        size_t o = ((size_t)b * T_SZ + t) * 2 + j;
        acc += pd[o];
        pp[o] = acc;
    }
}

// ---------------------------------------------------------------- launch
extern "C" void kernel_launch(void* const* d_in, const int* in_sizes, int n_in,
                              void* d_out, int out_size, void* d_ws, size_t ws_size,
                              hipStream_t stream) {
    const float* ev      = (const float*)d_in[0];  // B x D
    const float* teacher = (const float*)d_in[1];  // B x T x 2
    const float* w_init  = (const float*)d_in[2];  // H x D
    const float* b_init  = (const float*)d_in[3];  // H
    const float* w_ih    = (const float*)d_in[4];  // 3H x 2
    const float* w_hh    = (const float*)d_in[5];  // 3H x H
    const float* b_ih    = (const float*)d_in[6];  // 3H
    const float* b_hh    = (const float*)d_in[7];  // 3H
    const float* w_out   = (const float*)d_in[8];  // 2 x H
    const float* b_out   = (const float*)d_in[9];  // 2

    float* pd = (float*)d_out;                       // pred_deltas (B,T,2)
    float* pp = pd + (size_t)B_SZ * T_SZ * 2;        // pred_pos    (B,T,2)

    // workspace layout (bytes)
    char* ws = (char*)d_ws;
    u16*   evb  = (u16*)(ws);                                      //  8 MB
    u16*   wib  = (u16*)(ws + (size_t)8  * 1024 * 1024);           //  2 MB
    u16*   whhb = (u16*)(ws + (size_t)10 * 1024 * 1024);           //  6 MB
    float* hf0  = (float*)(ws + (size_t)16 * 1024 * 1024);         // 16 MB
    float* hf1  = (float*)(ws + (size_t)32 * 1024 * 1024);         // 16 MB
    u16*   hb0  = (u16*)(ws + (size_t)48 * 1024 * 1024);           //  8 MB
    u16*   hb1  = (u16*)(ws + (size_t)56 * 1024 * 1024);           //  8 MB
    float* hf[2] = {hf0, hf1};
    u16*   hb[2] = {hb0, hb1};

    // 1) fp32 -> bf16 conversions (weights once, ev once)
    {
        int n;
        n = B_SZ * D_SZ;
        cvt_f32_bf16_kernel<<<(n + 255) / 256, 256, 0, stream>>>(ev, evb, n);
        n = H_SZ * D_SZ;
        cvt_f32_bf16_kernel<<<(n + 255) / 256, 256, 0, stream>>>(w_init, wib, n);
        n = 3 * H_SZ * H_SZ;
        cvt_f32_bf16_kernel<<<(n + 255) / 256, 256, 0, stream>>>(w_hh, whhb, n);
    }

    // 2) h0 GEMM
    h0_kernel<<<dim3(B_SZ / 64, H_SZ / 32), 256, 0, stream>>>(
        evb, wib, b_init, hf[0], hb[0]);

    // 3) 50 sequential fused GRU steps (ping-pong h buffers) + per-step delta
    for (int t = 0; t < T_SZ; ++t) {
        int src = t & 1, dst = (t + 1) & 1;
        gru_step_kernel<<<dim3(B_SZ / 128, H_SZ / 32), 256, 0, stream>>>(
            hb[src], hf[src], whhb, b_hh, w_ih, b_ih, teacher, t,
            hf[dst], hb[dst]);
        delta_kernel<<<B_SZ / 8, 256, 0, stream>>>(hf[dst], w_out, b_out, pd, t);
    }

    // 4) cumsum -> pred_pos
    cumsum_kernel<<<(B_SZ * 2 + 255) / 256, 256, 0, stream>>>(pd, pp);
    (void)in_sizes; (void)n_in; (void)out_size; (void)ws_size;
}
